// AttentionDecoderRNN_60885456388843
// MI455X (gfx1250) — matrix operand order repaired and checked by reading, not compiled, not run on hardware
//
#include <hip/hip_runtime.h>
#include <cstdint>
#include <cstddef>

// ---------------------------------------------------------------------------
// Problem constants (match the JAX reference)
// ---------------------------------------------------------------------------
#define Bn   256
#define Sn   64
#define Hn   256
#define Vn   32000
#define Tn   11

typedef __attribute__((ext_vector_type(16))) __bf16 v16bf;
typedef __attribute__((ext_vector_type(8)))  __bf16 v8bf;
typedef __attribute__((ext_vector_type(8)))  float  v8f;

// ---------------------------------------------------------------------------
// Generic bf16 WMMA GEMM:  C[M,N] = A[M,K] @ Bw[N,K]^T + bias
// Bw is stored row-major [N,K] (== B-matrix columns are contiguous rows).
// Each wave computes a 32(M) x 64(N) strip: 2 A fragments x 4 B fragments,
// 8 WMMAs per K-step, each B fragment feeds two WMMAs.
// Optional tanh epilogue; optional bf16 mirror of C (same indexing).
// Requires: M%32==0, N%64==0, K%32==0.
// ---------------------------------------------------------------------------
__global__ void gemm_bf16_wmma(const __bf16* __restrict__ A,
                               const __bf16* __restrict__ Bw,
                               const float*  __restrict__ bias,
                               float*        __restrict__ C,
                               __bf16*       __restrict__ Cb,
                               int M, int N, int K, int ldc, int apply_tanh) {
  const int wave = (blockIdx.x * blockDim.x + threadIdx.x) >> 5;
  const int lane = threadIdx.x & 31;
  const int tilesM2 = M >> 5;          // 32-row strips
  const int ngroups = N >> 6;          // 64-col strips
  const int mt = wave % tilesM2;
  const int ng = wave / tilesM2;
  if (ng >= ngroups) return;

  // A-fragment addressing (16x32 bf16): lane L -> row L&15,
  // K chunks [kh*8 .. +7] and [16+kh*8 .. +7], kh = L>>4.
  const int akh  = (lane >> 4) << 3;               // 0 or 8
  const __bf16* arow0 = A + (size_t)((mt << 5) + (lane & 15)) * K;        // rows 0..15
  const __bf16* arow1 = A + (size_t)((mt << 5) + 16 + (lane & 15)) * K;   // rows 16..31

  // B-fragment addressing (32x16 bf16): lane L -> column L&15,
  // K chunks [kh*8 .. +7] and [16+kh*8 .. +7], kh = L>>4. Column n of B == row n of Bw.
  const int nb  = lane & 15;
  const int bkh = (lane >> 4) << 3;                // 0 or 8
  const int nbase = ng << 6;

  v8f acc0[4], acc1[4];
#pragma unroll
  for (int j = 0; j < 4; ++j) { acc0[j] = v8f{}; acc1[j] = v8f{}; }

  for (int k0 = 0; k0 < K; k0 += 32) {
    // Both A fragments loaded up front (clause together).
    v8bf a0lo = *(const v8bf*)(arow0 + k0 + akh);
    v8bf a0hi = *(const v8bf*)(arow0 + k0 + 16 + akh);
    v8bf a1lo = *(const v8bf*)(arow1 + k0 + akh);
    v8bf a1hi = *(const v8bf*)(arow1 + k0 + 16 + akh);
    v16bf a0 = __builtin_shufflevector(a0lo, a0hi,
        0,1,2,3,4,5,6,7,8,9,10,11,12,13,14,15);
    v16bf a1 = __builtin_shufflevector(a1lo, a1hi,
        0,1,2,3,4,5,6,7,8,9,10,11,12,13,14,15);
#pragma unroll
    for (int j = 0; j < 4; ++j) {
      const __bf16* bcol = Bw + (size_t)(nbase + (j << 4) + nb) * K + k0 + bkh;
      v8bf blo = *(const v8bf*)(bcol);
      v8bf bhi = *(const v8bf*)(bcol + 16);
      v16bf bfrag = __builtin_shufflevector(blo, bhi,
          0,1,2,3,4,5,6,7,8,9,10,11,12,13,14,15);
      acc0[j] = __builtin_amdgcn_wmma_f32_16x16x32_bf16(
          false, a0, false, bfrag, (short)0, acc0[j], false, false);
      acc1[j] = __builtin_amdgcn_wmma_f32_16x16x32_bf16(
          false, a1, false, bfrag, (short)0, acc1[j], false, false);
    }
  }

  // C/D layout: lane L holds column L&15, rows (L>>4)*8 + i  (i = vgpr idx)
  const int crow0 = (mt << 5) + ((lane >> 4) << 3);
#pragma unroll
  for (int j = 0; j < 4; ++j) {
    const int ncol = nbase + (j << 4) + nb;
    const float bs = bias ? bias[ncol] : 0.0f;
#pragma unroll
    for (int i = 0; i < 8; ++i) {
      float v = acc0[j][i] + bs;
      if (apply_tanh) v = tanhf(v);
      size_t idx = (size_t)(crow0 + i) * ldc + ncol;
      C[idx] = v;
      if (Cb) Cb[idx] = (__bf16)v;

      float v1 = acc1[j][i] + bs;
      if (apply_tanh) v1 = tanhf(v1);
      size_t idx1 = (size_t)(crow0 + 16 + i) * ldc + ncol;
      C[idx1] = v1;
      if (Cb) Cb[idx1] = (__bf16)v1;
    }
  }
}

// ---------------------------------------------------------------------------
// Fused attention step: scores -> softmax -> ctx, writes attn into d_out and
// builds bf16 z = [emb, ctx, h] for the GRU GEMM.  One block per batch row.
// ---------------------------------------------------------------------------
__global__ void attention_step(const float* __restrict__ q,        // [B,H]
                               const float* __restrict__ uk,       // [B,S,H]
                               const float* __restrict__ Va,       // [H]
                               const float* __restrict__ bva,      // [1]
                               const float* __restrict__ enc,      // [B,S,H]
                               const float* __restrict__ h,        // [B,H]
                               const float* __restrict__ embedding,// [V,H]
                               const int*   __restrict__ target,   // [B,T]
                               __bf16*      __restrict__ z,        // [B,3H]
                               float*       __restrict__ out_att,  // [B,T,S]
                               int t) {
  const int b = blockIdx.x, tid = threadIdx.x;
  __shared__ float q_sh[Hn], va_sh[Hn], sc[Sn];
  __shared__ float inv_sh;
  q_sh[tid]  = q[b * Hn + tid];
  va_sh[tid] = Va[tid];
  __syncthreads();

  const int wv = tid >> 5, lane = tid & 31;
  const float* ukb = uk + (size_t)b * Sn * Hn;
  for (int s = wv; s < Sn; s += 8) {
    float acc = 0.0f;
    for (int hh = lane; hh < Hn; hh += 32)
      acc += tanhf(q_sh[hh] + ukb[s * Hn + hh]) * va_sh[hh];
#pragma unroll
    for (int o = 16; o; o >>= 1) acc += __shfl_xor(acc, o, 32);
    if (lane == 0) sc[s] = acc + bva[0];
  }
  __syncthreads();

  if (tid == 0) {
    float mx = -INFINITY;
    for (int s = 0; s < Sn; ++s) mx = fmaxf(mx, sc[s]);
    float sum = 0.0f;
    for (int s = 0; s < Sn; ++s) { float e = expf(sc[s] - mx); sc[s] = e; sum += e; }
    inv_sh = 1.0f / sum;
  }
  __syncthreads();
  if (tid < Sn) {
    float a = sc[tid] * inv_sh;
    sc[tid] = a;
    out_att[(size_t)b * (Tn * Sn) + t * Sn + tid] = a;
  }
  __syncthreads();

  // ctx[b, tid] = sum_s attn[s] * enc[b, s, tid]
  float ctx = 0.0f;
  const float* encb = enc + (size_t)b * Sn * Hn;
  for (int s = 0; s < Sn; ++s) ctx += sc[s] * encb[s * Hn + tid];

  const int tok = (t == 0) ? 0 : target[b * Tn + (t - 1)];
  __bf16* zb = z + (size_t)b * (3 * Hn);
  zb[tid]            = (__bf16)embedding[(size_t)tok * Hn + tid];
  zb[Hn + tid]       = (__bf16)ctx;
  zb[2 * Hn + tid]   = (__bf16)h[b * Hn + tid];
}

// ---------------------------------------------------------------------------
// Log-softmax over rows of length Vn: pass 1 streaming log-sum-exp, pass 2
// in-place normalize.
// ---------------------------------------------------------------------------
__global__ void row_lse(const float* __restrict__ X, float* __restrict__ lse, int n) {
  const int r = blockIdx.x;
  const float* x = X + (size_t)r * n;
  float m = -INFINITY, s = 0.0f;
  for (int i = threadIdx.x; i < n; i += blockDim.x) {
    float v = x[i];
    if (v > m) { s = s * expf(m - v) + 1.0f; m = v; }
    else       { s += expf(v - m); }
  }
  __shared__ float sm[256], ss[256];
  sm[threadIdx.x] = m; ss[threadIdx.x] = s;
  __syncthreads();
  if (threadIdx.x == 0) {
    float M = -INFINITY, S = 0.0f;
    for (int i = 0; i < 256; ++i) {
      float mi = sm[i], si = ss[i];
      if (mi > M) { S = S * expf(M - mi) + si; M = mi; }
      else        { S += si * expf(mi - M); }
    }
    lse[r] = M + logf(S);
  }
}

__global__ void row_normalize(float* __restrict__ X, const float* __restrict__ lse, int n) {
  const int r = blockIdx.x;
  const float L = lse[r];
  float* x = X + (size_t)r * n;
  for (int i = threadIdx.x; i < n; i += blockDim.x) x[i] -= L;
}

// ---------------------------------------------------------------------------
// Preprocessing helpers
// ---------------------------------------------------------------------------
__global__ void f32_to_bf16(const float* __restrict__ s, __bf16* __restrict__ d, int n) {
  for (int i = blockIdx.x * blockDim.x + threadIdx.x; i < n; i += gridDim.x * blockDim.x)
    d[i] = (__bf16)s[i];
}

// Wcat[n, 0:2H] = W_ih[n,:]; Wcat[n, 2H:3H] = W_hh[n,:]
__global__ void build_wcat(const float* __restrict__ W_ih, const float* __restrict__ W_hh,
                           __bf16* __restrict__ Wcat) {
  const int n = Hn * 3 * Hn;
  for (int i = blockIdx.x * blockDim.x + threadIdx.x; i < n; i += gridDim.x * blockDim.x) {
    int row = i / (3 * Hn), c = i % (3 * Hn);
    float v = (c < 2 * Hn) ? W_ih[row * 2 * Hn + c] : W_hh[row * Hn + (c - 2 * Hn)];
    Wcat[i] = (__bf16)v;
  }
}

__global__ void combine_bias(const float* a, const float* b, float* out) {
  int i = threadIdx.x;
  out[i] = a[i] + b[i];
}

__global__ void init_h(const float* __restrict__ eh, float* __restrict__ h,
                       __bf16* __restrict__ hb) {
  int i = blockIdx.x * blockDim.x + threadIdx.x;   // B*H threads
  float v = eh[i];
  h[i] = v;
  hb[i] = (__bf16)v;
}

// ---------------------------------------------------------------------------
// Host launcher
// ---------------------------------------------------------------------------
extern "C" void kernel_launch(void* const* d_in, const int* in_sizes, int n_in,
                              void* d_out, int out_size, void* d_ws, size_t ws_size,
                              hipStream_t stream) {
  const float* enc    = (const float*)d_in[0];   // [B,S,H]
  const float* enc_h  = (const float*)d_in[1];   // [1,B,H]
  const int*   target = (const int*)  d_in[2];   // [B,T]
  const float* embed  = (const float*)d_in[3];   // [V,H]
  const float* Wa     = (const float*)d_in[4];
  const float* ba     = (const float*)d_in[5];
  const float* Ua     = (const float*)d_in[6];
  const float* bua    = (const float*)d_in[7];
  const float* Va     = (const float*)d_in[8];
  const float* bva    = (const float*)d_in[9];
  const float* W_ih   = (const float*)d_in[10];
  const float* b_ih   = (const float*)d_in[11];
  const float* W_hh   = (const float*)d_in[12];
  const float* b_hh   = (const float*)d_in[13];
  const float* W_out  = (const float*)d_in[14];
  const float* b_out  = (const float*)d_in[15];

  float* out_dec = (float*)d_out;                       // [B,T,V]
  float* out_h   = out_dec + (size_t)Bn * Tn * Vn;      // [1,B,H]
  float* out_att = out_h + (size_t)Bn * Hn;             // [B,T,S]

  // Workspace carving (256B aligned slabs)
  uint8_t* w = (uint8_t*)d_ws;
  size_t off = 0;
  auto take = [&](size_t bytes) -> void* {
    void* p = w + off;
    off = (off + bytes + 255) & ~(size_t)255;
    return p;
  };
  float*  uk      = (float*) take((size_t)Bn * Sn * Hn * 4);     // 16.8 MB
  float*  qbuf    = (float*) take((size_t)Bn * Hn * 4);
  float*  hbuf    = (float*) take((size_t)Bn * Hn * 4);
  float*  lse     = (float*) take((size_t)Bn * Tn * 4);
  float*  bcomb   = (float*) take((size_t)Hn * 4);
  __bf16* enc_bf  = (__bf16*)take((size_t)Bn * Sn * Hn * 2);     // 8.4 MB
  __bf16* Wa_bf   = (__bf16*)take((size_t)Hn * Hn * 2);
  __bf16* Ua_bf   = (__bf16*)take((size_t)Hn * Hn * 2);
  __bf16* Wout_bf = (__bf16*)take((size_t)Vn * Hn * 2);          // 16.4 MB
  __bf16* Wcat_bf = (__bf16*)take((size_t)Hn * 3 * Hn * 2);
  __bf16* z_bf    = (__bf16*)take((size_t)Bn * 3 * Hn * 2);
  __bf16* h_bf    = (__bf16*)take((size_t)Bn * Hn * 2);

  // ---- preprocessing ----
  f32_to_bf16<<<512, 256, 0, stream>>>(enc,   enc_bf,  Bn * Sn * Hn);
  f32_to_bf16<<<64,  256, 0, stream>>>(Wa,    Wa_bf,   Hn * Hn);
  f32_to_bf16<<<64,  256, 0, stream>>>(Ua,    Ua_bf,   Hn * Hn);
  f32_to_bf16<<<1024,256, 0, stream>>>(W_out, Wout_bf, Vn * Hn);
  build_wcat<<<192, 256, 0, stream>>>(W_ih, W_hh, Wcat_bf);
  combine_bias<<<1, Hn, 0, stream>>>(b_ih, b_hh, bcomb);
  init_h<<<(Bn * Hn) / 256, 256, 0, stream>>>(enc_h, hbuf, h_bf);

  // uk = enc @ Ua^T + bua   [B*S, H]
  {
    int waves = ((Bn * Sn) / 32) * (Hn / 64);        // 2048
    gemm_bf16_wmma<<<(waves + 7) / 8, 256, 0, stream>>>(
        enc_bf, Ua_bf, bua, uk, nullptr, Bn * Sn, Hn, Hn, Hn, 0);
  }

  const int waves_q   = (Bn / 32) * (Hn / 64);       // 32
  const int waves_out = (Bn / 32) * (Vn / 64);       // 4000

  for (int t = 0; t < Tn; ++t) {
    // q = h @ Wa^T + ba
    gemm_bf16_wmma<<<(waves_q + 7) / 8, 256, 0, stream>>>(
        h_bf, Wa_bf, ba, qbuf, nullptr, Bn, Hn, Hn, Hn, 0);
    // attention + softmax + ctx + build z (also writes attentions to d_out)
    attention_step<<<Bn, Hn, 0, stream>>>(
        qbuf, uk, Va, bva, enc, hbuf, embed, target, z_bf, out_att, t);
    // h_new = tanh(z @ Wcat^T + (b_ih + b_hh)), also emit bf16 copy
    gemm_bf16_wmma<<<(waves_q + 7) / 8, 256, 0, stream>>>(
        z_bf, Wcat_bf, bcomb, hbuf, h_bf, Bn, Hn, 3 * Hn, Hn, 1);
    // logits -> d_out slot for step t (rows strided by T*V)
    gemm_bf16_wmma<<<(waves_out + 7) / 8, 256, 0, stream>>>(
        h_bf, Wout_bf, b_out, out_dec + (size_t)t * Vn, nullptr,
        Bn, Vn, Hn, Tn * Vn, 0);
  }

  // log_softmax in place over the B*T rows of length V
  row_lse<<<Bn * Tn, 256, 0, stream>>>(out_dec, lse, Vn);
  row_normalize<<<Bn * Tn, 256, 0, stream>>>(out_dec, lse, Vn);

  // h_last
  hipMemcpyAsync(out_h, hbuf, (size_t)Bn * Hn * sizeof(float),
                 hipMemcpyDeviceToDevice, stream);
  (void)in_sizes; (void)n_in; (void)out_size; (void)ws_size;
}
